// SimpleTransformerBlock_88330297409580
// MI455X (gfx1250) — matrix operand order repaired and checked by reading, not compiled, not run on hardware
//
#include <hip/hip_runtime.h>

// ---------------------------------------------------------------------------
// SimpleTransformerBlock for MI455X (gfx1250), bf16 WMMA with fp32 accumulate.
// B=2 T=2048 D=1024 H=16 HD=64.  All matmuls use v_wmma_f32_16x16x32_bf16;
// global->LDS staging uses GLOBAL_LOAD_ASYNC_TO_LDS_B128 when available.
// ---------------------------------------------------------------------------

#define B_   2
#define T_   2048
#define D_   1024
#define H_   16
#define HD_  64
#define MTOK (B_ * T_)   // 4096 token rows

typedef __attribute__((ext_vector_type(16))) __bf16       bf16x16;
typedef __attribute__((ext_vector_type(8)))  float        f32x8;
typedef __attribute__((ext_vector_type(4)))  unsigned int u32x4;
typedef __attribute__((ext_vector_type(4)))  int          i32x4;

union FragBF16 {
    bf16x16        v;
    u32x4          q[2];
    unsigned short h[16];
};

__device__ __forceinline__ unsigned short f32_to_bf16(float f) {
    unsigned u = __float_as_uint(f);
    unsigned r = 0x7FFFu + ((u >> 16) & 1u);   // round-to-nearest-even
    return (unsigned short)((u + r) >> 16);
}

// ---- async global->LDS b128 copy (CDNA5), with synchronous fallback -------
#if defined(__has_builtin)
#if __has_builtin(__builtin_amdgcn_global_load_async_to_lds_b128)
#define ASYNC_LDS_COPY 1
#endif
#if __has_builtin(__builtin_amdgcn_s_wait_asynccnt)
#define HAVE_WAIT_ASYNC_BUILTIN 1
#endif
#endif

__device__ __forceinline__ void lds_copy_b128(unsigned short* dst_lds,
                                              const unsigned short* src_glb) {
#ifdef ASYNC_LDS_COPY
    // builtin signature (from hipcc diagnostic): (int4 AS1*, int4 AS3*, Ii, Ii)
    typedef __attribute__((address_space(1))) i32x4* glb_p;
    typedef __attribute__((address_space(3))) i32x4* lds_p;
    __builtin_amdgcn_global_load_async_to_lds_b128(
        (glb_p)(unsigned long long)src_glb,
        (lds_p)(unsigned)(unsigned long long)dst_lds, 0, 0);
#else
    *(u32x4*)dst_lds = *(const u32x4*)src_glb;
#endif
}

__device__ __forceinline__ void lds_copy_wait() {
#ifdef ASYNC_LDS_COPY
#ifdef HAVE_WAIT_ASYNC_BUILTIN
    __builtin_amdgcn_s_wait_asynccnt(0);
#else
    asm volatile("s_wait_asynccnt 0" ::: "memory");
#endif
#endif
}

// ---------------------------------------------------------------------------
// One-time weight prep: fp32 [K][N] -> bf16 transposed [N][K] (tiled in LDS
// so both global sides are coalesced).  Amortized over 32 GEMM M-blocks.
// ---------------------------------------------------------------------------
__global__ __launch_bounds__(256)
void transpose_f32_to_bf16_kernel(const float* __restrict__ in,   // [K][N]
                                  unsigned short* __restrict__ out, // [N][K]
                                  int K, int N) {
    __shared__ unsigned short tile[32][33];
    const int k0 = blockIdx.y * 32;
    const int n0 = blockIdx.x * 32;
    const int c  = threadIdx.x & 31;
    const int r0 = (threadIdx.x >> 5) * 4;
#pragma unroll
    for (int i = 0; i < 4; ++i) {
        const int r = r0 + i;
        tile[r][c] = f32_to_bf16(in[(size_t)(k0 + r) * N + n0 + c]);
    }
    __syncthreads();
#pragma unroll
    for (int i = 0; i < 4; ++i) {
        const int r = r0 + i;
        out[(size_t)(n0 + r) * K + k0 + c] = tile[c][r];
    }
}

// ---------------------------------------------------------------------------
// One-time V prep: gather V out of the packed qkv (bf16) and transpose per
// (b,h):  vt[(b*H+h)*HD + d][t] = qkv[b*T+t][2*D + h*HD + d]
// ---------------------------------------------------------------------------
__global__ __launch_bounds__(256)
void transpose_v_kernel(const unsigned short* __restrict__ qkv,
                        unsigned short* __restrict__ vt) {
    __shared__ unsigned short tile[32][33];
    const int bh = blockIdx.z;                  // b*H + h
    const int b  = bh >> 4, h = bh & 15;
    const int t0 = blockIdx.x * 32;
    const int d0 = blockIdx.y * 32;
    const int c  = threadIdx.x & 31;
    const int r0 = (threadIdx.x >> 5) * 4;
    const unsigned short* vin =
        qkv + (size_t)b * T_ * 3 * D_ + 2 * D_ + h * HD_;
#pragma unroll
    for (int i = 0; i < 4; ++i) {
        const int r = r0 + i;
        tile[r][c] = vin[(size_t)(t0 + r) * 3 * D_ + d0 + c];
    }
    __syncthreads();
    unsigned short* vo = vt + ((size_t)bh * HD_ + d0) * T_ + t0;
#pragma unroll
    for (int i = 0; i < 4; ++i) {
        const int r = r0 + i;
        vo[(size_t)r * T_ + c] = tile[c][r];
    }
}

// ---------------------------------------------------------------------------
// LayerNorm over D=1024, one workgroup (256 threads = 8 waves) per row.
// ---------------------------------------------------------------------------
__global__ __launch_bounds__(256)
void layernorm_kernel(const float* __restrict__ x,
                      const float* __restrict__ g,
                      const float* __restrict__ be,
                      unsigned short* __restrict__ out) {
    __shared__ float ssum[256];
    __shared__ float ssq[256];
    const int tid = threadIdx.x;
    const size_t row = blockIdx.x;
    const float* xr = x + row * D_;

    float lx[4];
    float s = 0.f, sq = 0.f;
#pragma unroll
    for (int i = 0; i < 4; ++i) {
        float v = xr[tid + 256 * i];
        lx[i] = v; s += v; sq += v * v;
    }
    ssum[tid] = s; ssq[tid] = sq;
    __syncthreads();
    for (int o = 128; o > 0; o >>= 1) {
        if (tid < o) { ssum[tid] += ssum[tid + o]; ssq[tid] += ssq[tid + o]; }
        __syncthreads();
    }
    const float mu  = ssum[0] * (1.f / D_);
    const float var = ssq[0] * (1.f / D_) - mu * mu;
    const float rs  = rsqrtf(var + 1e-5f);
#pragma unroll
    for (int i = 0; i < 4; ++i) {
        int c = tid + 256 * i;
        out[row * D_ + c] = f32_to_bf16((lx[i] - mu) * rs * g[c] + be[c]);
    }
}

// ---------------------------------------------------------------------------
// Tiled bf16 WMMA GEMM:  C[M,N] = A[M,K] @ Bt^T   (Bt is bf16 [N][K])
// Block 128x128, K-step 64, 256 threads = 8 waves (4 M x 2 N), each wave
// computes 32x64 = eight 16x16 fp32 accumulators (16 WMMA / K-step).
// Double-buffered LDS (2 x 36KB); staging is pure b128 global->LDS copies
// (async when available), prefetching tile i+1 during compute of tile i.
// 144B LDS rows keep 16B alignment so every fragment gather is ds_load_b128.
// ---------------------------------------------------------------------------
#define EPI_BF16     0
#define EPI_RES_F32  1
#define EPI_GELU     2

template <int EPI>
__global__ __launch_bounds__(256)
void gemm_bf16_kernel(const unsigned short* __restrict__ A,   // [M][K]
                      const unsigned short* __restrict__ Bt,  // [N][K]
                      int M, int N, int K,
                      const float* __restrict__ bias,
                      const float* __restrict__ residual,
                      float* __restrict__ outF,
                      unsigned short* __restrict__ outB) {
    __shared__ __align__(16) unsigned short As[2][128][72];
    __shared__ __align__(16) unsigned short Bs[2][128][72];   // [n][k]

    const int tid  = threadIdx.x;
    const int lane = tid & 31;
    const int wid  = tid >> 5;
    const int wm   = wid & 3;        // wave position along M (4)
    const int wn   = wid >> 2;       // wave position along N (2)
    const int lm   = lane & 15;
    const int lh   = lane >> 4;      // lane half (0/1)
    const int m0   = blockIdx.y * 128;
    const int n0   = blockIdx.x * 128;

    f32x8 acc[2][4];
#pragma unroll
    for (int mi = 0; mi < 2; ++mi)
#pragma unroll
        for (int ni = 0; ni < 4; ++ni)
#pragma unroll
            for (int v = 0; v < 8; ++v) acc[mi][ni][v] = 0.f;

    // cooperative staging: thread t moves row t/2, 32-elem chunk (t&1)*32
    const int sr = tid >> 1;
    const int sc = (tid & 1) * 32;
    const unsigned short* arow = A  + (size_t)(m0 + sr) * K + sc;
    const unsigned short* brow = Bt + (size_t)(n0 + sr) * K + sc;

    const int nk = K / 64;
    // prologue: stage tile 0
#pragma unroll
    for (int c = 0; c < 4; ++c) {
        lds_copy_b128(&As[0][sr][sc + 8 * c], arow + 8 * c);
        lds_copy_b128(&Bs[0][sr][sc + 8 * c], brow + 8 * c);
    }
    lds_copy_wait();
    __syncthreads();

    for (int kt = 0; kt < nk; ++kt) {
        const int buf = kt & 1;
        if (kt + 1 < nk) {   // prefetch next K-tile into the other buffer
            const size_t ko = (size_t)(kt + 1) * 64;
#pragma unroll
            for (int c = 0; c < 4; ++c) {
                lds_copy_b128(&As[buf ^ 1][sr][sc + 8 * c], arow + ko + 8 * c);
                lds_copy_b128(&Bs[buf ^ 1][sr][sc + 8 * c], brow + ko + 8 * c);
            }
        }

        // fragment gathers per ISA 16-bit layouts (all ds_load_b128)
        FragBF16 af[2][2], bfr[2][4];
#pragma unroll
        for (int kk = 0; kk < 2; ++kk) {
#pragma unroll
            for (int mi = 0; mi < 2; ++mi) {
                const unsigned short* p =
                    &As[buf][wm * 32 + mi * 16 + lm][kk * 32 + lh * 8];
                af[kk][mi].q[0] = *(const u32x4*)p;          // K = +0..7
                af[kk][mi].q[1] = *(const u32x4*)(p + 16);   // K = +16..23
            }
#pragma unroll
            for (int ni = 0; ni < 4; ++ni) {
                const unsigned short* p =
                    &Bs[buf][wn * 64 + ni * 16 + lm][kk * 32 + lh * 8];
                bfr[kk][ni].q[0] = *(const u32x4*)p;         // K = +0..7
                bfr[kk][ni].q[1] = *(const u32x4*)(p + 16);  // K = +16..23
            }
        }
#pragma unroll
        for (int kk = 0; kk < 2; ++kk)
#pragma unroll
            for (int mi = 0; mi < 2; ++mi)
#pragma unroll
                for (int ni = 0; ni < 4; ++ni)
                    acc[mi][ni] = __builtin_amdgcn_wmma_f32_16x16x32_bf16(
                        false, af[kk][mi].v, false, bfr[kk][ni].v, (short)0,
                        acc[mi][ni], false, false);
        lds_copy_wait();
        __syncthreads();
    }

    // epilogue: C layout lane l, vgpr v -> row v+8*(l>=16), col l%16
#pragma unroll
    for (int mi = 0; mi < 2; ++mi) {
#pragma unroll
        for (int ni = 0; ni < 4; ++ni) {
            const int col = n0 + wn * 64 + ni * 16 + lm;
            const float bv = (EPI != EPI_BF16 && bias) ? bias[col] : 0.f;
#pragma unroll
            for (int v = 0; v < 8; ++v) {
                const int row = m0 + wm * 32 + mi * 16 + v + 8 * lh;
                const size_t idx = (size_t)row * N + col;
                const float val = acc[mi][ni][v];
                if (EPI == EPI_BF16) {
                    outB[idx] = f32_to_bf16(val);
                } else if (EPI == EPI_RES_F32) {
                    outF[idx] = residual[idx] + val + bv;
                } else {  // bias + exact-erf GELU, bf16 out
                    const float xg = val + bv;
                    const float gl = 0.5f * xg * (1.f + erff(xg * 0.70710678f));
                    outB[idx] = f32_to_bf16(gl);
                }
            }
        }
    }
}

// ---------------------------------------------------------------------------
// Flash attention.  One workgroup = 4 waves sharing (b,h); wave w owns the
// 16-row q-tile (qblk*4 + w).  Keys streamed in blocks of 32: K (row-major)
// and pre-transposed V tiles staged in LDS via b128 copies (async when
// available), so S = Q@K^T and O += P@V fragment gathers are all
// ds_load_b128.  Online softmax via 16-lane shfl_xor reductions (wave32
// halves match the C-matrix row split).
// ---------------------------------------------------------------------------
__global__ __launch_bounds__(128)
void attention_kernel(const unsigned short* __restrict__ qkv,
                      const unsigned short* __restrict__ vtg,  // [BH][HD][T]
                      unsigned short* __restrict__ attnout) {
    __shared__ __align__(16) unsigned short Kt[32][72];       // [key][dim]
    __shared__ __align__(16) unsigned short Vt[64][40];       // [dim][key]
    __shared__ __align__(16) unsigned short Plds[4][16][40];  // per-wave P

    const int tid  = threadIdx.x;
    const int lane = tid & 31;
    const int wid  = tid >> 5;
    const int qblk = blockIdx.x & 31;            // 32 q-blocks of 64 rows
    const int h    = (blockIdx.x >> 5) & 15;
    const int b    = blockIdx.x >> 9;
    const int lm   = lane & 15;
    const int lh   = lane >> 4;

    const int RS = 3 * D_;                       // 3072 token row stride
    const size_t tok0 = (size_t)b * T_;
    const unsigned short* qbase = qkv + tok0 * RS + h * HD_;
    const unsigned short* kbase = qbase + D_;
    const unsigned short* vbase = vtg + (size_t)(b * H_ + h) * HD_ * T_;

    const int qt = qblk * 4 + wid;               // this wave's 16-row q tile

    // Q A-fragments (16x64 -> two 16x32 frags), loaded once
    FragBF16 qa[2];
    {
        const unsigned short* qp = qbase + (size_t)(qt * 16 + lm) * RS;
#pragma unroll
        for (int kk = 0; kk < 2; ++kk) {
            const int c0 = kk * 32 + (lh ? 8 : 0);
            qa[kk].q[0] = *(const u32x4*)(qp + c0);
            qa[kk].q[1] = *(const u32x4*)(qp + c0 + 16);
        }
    }

    f32x8 o[4];
    float m[8], lsum[8];
#pragma unroll
    for (int f = 0; f < 4; ++f)
#pragma unroll
        for (int v = 0; v < 8; ++v) o[f][v] = 0.f;
#pragma unroll
    for (int v = 0; v < 8; ++v) { m[v] = -INFINITY; lsum[v] = 0.f; }

    const float scale = 0.125f;   // 1/sqrt(64)

    // staging assignments (128 threads):
    const int lkey = tid & 31;            // K: key row, chunk (tid>>5)*16
    const int lkc  = (tid >> 5) * 16;
    const int lvd  = tid >> 1;            // V: dim row 0..63
    const int lvc  = (tid & 1) * 16;      // V: key chunk {0,16}

    for (int kp = 0; kp < T_; kp += 32) {
        // ---- stage K (row-major) and pre-transposed V tiles in LDS ----
        {
            const unsigned short* kr = kbase + (size_t)(kp + lkey) * RS + lkc;
            lds_copy_b128(&Kt[lkey][lkc],     kr);
            lds_copy_b128(&Kt[lkey][lkc + 8], kr + 8);
            const unsigned short* vr = vbase + (size_t)lvd * T_ + kp + lvc;
            lds_copy_b128(&Vt[lvd][lvc],     vr);
            lds_copy_b128(&Vt[lvd][lvc + 8], vr + 8);
        }
        lds_copy_wait();
        __syncthreads();

        // ---- S = Q @ K^T for 32 keys (two 16x16 tiles) ----
        f32x8 s[2];
#pragma unroll
        for (int half = 0; half < 2; ++half) {
            f32x8 accS;
#pragma unroll
            for (int v = 0; v < 8; ++v) accS[v] = 0.f;
#pragma unroll
            for (int kk = 0; kk < 2; ++kk) {
                FragBF16 kb;
                const unsigned short* p = &Kt[half * 16 + lm][kk * 32 + lh * 8];
                kb.q[0] = *(const u32x4*)p;
                kb.q[1] = *(const u32x4*)(p + 16);
                accS = __builtin_amdgcn_wmma_f32_16x16x32_bf16(
                    false, qa[kk].v, false, kb.v, (short)0, accS, false, false);
            }
            s[half] = accS;
        }

        // ---- online softmax over the 32 new columns ----
        float alpha[8];
#pragma unroll
        for (int v = 0; v < 8; ++v) {
            float s0 = s[0][v] * scale;
            float s1 = s[1][v] * scale;
            float mx = fmaxf(s0, s1);
#pragma unroll
            for (int ofs = 1; ofs < 16; ofs <<= 1)
                mx = fmaxf(mx, __shfl_xor(mx, ofs));
            const float mnew = fmaxf(m[v], mx);
            const float p0 = __expf(s0 - mnew);
            const float p1 = __expf(s1 - mnew);
            s[0][v] = p0; s[1][v] = p1;
            float rs = p0 + p1;
#pragma unroll
            for (int ofs = 1; ofs < 16; ofs <<= 1)
                rs += __shfl_xor(rs, ofs);
            const float a = __expf(m[v] - mnew);
            lsum[v] = lsum[v] * a + rs;
            m[v] = mnew;
            alpha[v] = a;
        }
#pragma unroll
        for (int f = 0; f < 4; ++f)
#pragma unroll
            for (int v = 0; v < 8; ++v) o[f][v] *= alpha[v];

        // ---- P (C-layout) -> LDS -> A-layout bf16 fragment ----
        unsigned short (*P)[40] = Plds[wid];
#pragma unroll
        for (int v = 0; v < 8; ++v) {
            const int r = v + 8 * lh;
            P[r][lm]      = f32_to_bf16(s[0][v]);
            P[r][16 + lm] = f32_to_bf16(s[1][v]);
        }
        FragBF16 pa;
        {
            const unsigned short* pr = &P[lm][lh * 8];
            pa.q[0] = *(const u32x4*)pr;
            pa.q[1] = *(const u32x4*)(pr + 16);
        }

        // ---- O += P(16x32) @ V(32x64), V fragments from transposed LDS ----
#pragma unroll
        for (int f = 0; f < 4; ++f) {
            FragBF16 vb;
            const unsigned short* p = &Vt[f * 16 + lm][lh * 8];
            vb.q[0] = *(const u32x4*)p;
            vb.q[1] = *(const u32x4*)(p + 16);
            o[f] = __builtin_amdgcn_wmma_f32_16x16x32_bf16(
                false, pa.v, false, vb.v, (short)0, o[f], false, false);
        }
        __syncthreads();   // before next iteration overwrites Kt/Vt
    }

    // ---- normalize and store bf16 [B*T][D] ----
    unsigned short* ob = attnout + (tok0 + (size_t)qt * 16) * D_ + h * HD_;
#pragma unroll
    for (int v = 0; v < 8; ++v) {
        const float inv = 1.f / lsum[v];
        const int row = v + 8 * lh;
#pragma unroll
        for (int f = 0; f < 4; ++f)
            ob[(size_t)row * D_ + f * 16 + lm] = f32_to_bf16(o[f][v] * inv);
    }
}

// ---------------------------------------------------------------------------
// host launcher
// ---------------------------------------------------------------------------
extern "C" void kernel_launch(void* const* d_in, const int* in_sizes, int n_in,
                              void* d_out, int out_size, void* d_ws,
                              size_t ws_size, hipStream_t stream) {
    (void)in_sizes; (void)n_in; (void)out_size; (void)ws_size;
    const float* x     = (const float*)d_in[0];
    const float* w_qkv = (const float*)d_in[1];
    const float* w_out = (const float*)d_in[2];
    const float* w_ff1 = (const float*)d_in[3];
    const float* b_ff1 = (const float*)d_in[4];
    const float* w_ff2 = (const float*)d_in[5];
    const float* b_ff2 = (const float*)d_in[6];
    const float* g1    = (const float*)d_in[7];
    const float* be1   = (const float*)d_in[8];
    const float* g2    = (const float*)d_in[9];
    const float* be2   = (const float*)d_in[10];
    float* out = (float*)d_out;

    char* ws = (char*)d_ws;
    size_t off = 0;
    auto take = [&](size_t bytes) -> char* {
        char* p = ws + off;
        off += (bytes + 255) & ~(size_t)255;
        return p;
    };
    // all weights stored transposed: [N][K] bf16
    unsigned short* wqkv_t = (unsigned short*)take((size_t)D_ * 3 * D_ * 2);
    unsigned short* wout_t = (unsigned short*)take((size_t)D_ * D_ * 2);
    unsigned short* wff1_t = (unsigned short*)take((size_t)D_ * 4 * D_ * 2);
    unsigned short* wff2_t = (unsigned short*)take((size_t)4 * D_ * D_ * 2);
    unsigned short* xn_b   = (unsigned short*)take((size_t)MTOK * D_ * 2);
    unsigned short* qkv_b  = (unsigned short*)take((size_t)MTOK * 3 * D_ * 2);
    unsigned short* vt_b   = (unsigned short*)take((size_t)B_ * H_ * HD_ * T_ * 2);
    unsigned short* att_b  = (unsigned short*)take((size_t)MTOK * D_ * 2);
    float*          x1     = (float*)take((size_t)MTOK * D_ * 4);
    unsigned short* h_b    = (unsigned short*)take((size_t)MTOK * 4 * D_ * 2);

    // weight conversion fp32 [K][N] -> bf16 transposed [N][K]
    transpose_f32_to_bf16_kernel<<<dim3(3 * D_ / 32, D_ / 32), 256, 0, stream>>>(
        w_qkv, wqkv_t, D_, 3 * D_);
    transpose_f32_to_bf16_kernel<<<dim3(D_ / 32, D_ / 32), 256, 0, stream>>>(
        w_out, wout_t, D_, D_);
    transpose_f32_to_bf16_kernel<<<dim3(4 * D_ / 32, D_ / 32), 256, 0, stream>>>(
        w_ff1, wff1_t, D_, 4 * D_);
    transpose_f32_to_bf16_kernel<<<dim3(D_ / 32, 4 * D_ / 32), 256, 0, stream>>>(
        w_ff2, wff2_t, 4 * D_, D_);

    // LN1 -> QKV -> V transpose -> attention -> out-proj (+residual)
    layernorm_kernel<<<MTOK, 256, 0, stream>>>(x, g1, be1, xn_b);
    gemm_bf16_kernel<EPI_BF16><<<dim3(3 * D_ / 128, MTOK / 128), 256, 0, stream>>>(
        xn_b, wqkv_t, MTOK, 3 * D_, D_, nullptr, nullptr, nullptr, qkv_b);
    transpose_v_kernel<<<dim3(T_ / 32, HD_ / 32, B_ * H_), 256, 0, stream>>>(
        qkv_b, vt_b);
    attention_kernel<<<B_ * H_ * (T_ / 64), 128, 0, stream>>>(qkv_b, vt_b, att_b);
    gemm_bf16_kernel<EPI_RES_F32><<<dim3(D_ / 128, MTOK / 128), 256, 0, stream>>>(
        att_b, wout_t, MTOK, D_, D_, nullptr, x, x1, nullptr);

    // LN2 -> FF1(+GELU) -> FF2 (+bias +residual) -> d_out
    layernorm_kernel<<<MTOK, 256, 0, stream>>>(x1, g2, be2, xn_b);
    gemm_bf16_kernel<EPI_GELU><<<dim3(4 * D_ / 128, MTOK / 128), 256, 0, stream>>>(
        xn_b, wff1_t, MTOK, 4 * D_, D_, b_ff1, nullptr, nullptr, h_b);
    gemm_bf16_kernel<EPI_RES_F32><<<dim3(D_ / 128, MTOK / 128), 256, 0, stream>>>(
        h_b, wff2_t, MTOK, D_, 4 * D_, b_ff2, x1, out, nullptr);
}
